// ModelParallelMultiheadAttention_71047349010595
// MI455X (gfx1250) — matrix operand order repaired and checked by reading, not compiled, not run on hardware
//
#include <hip/hip_runtime.h>
#include <cstdint>
#include <cstddef>

// ---------------------------------------------------------------------------
// MI455X (gfx1250) fused multi-head attention, bf16 WMMA + TDM path.
//   1) cvt: f32 -> bf16 for query, qkv_w, out_w
//   2) gemm<0>: KVQ = X*Wqkv^T + b, tiles staged by tensor_load_to_lds (TDM),
//               fused split/scale/relayout to Q,K (b,h,t,d), V^T (b,h,d,t)
//   3) flash_attn: per-(b,h,128-row) causal online-softmax attention, WMMA for
//               QK^T and P*V, P staged via wave-private LDS into A-frag layout
//   4) gemm<1>: out = ctx*Wout^T + b -> f32 d_out
// ---------------------------------------------------------------------------

#define DEVINL __device__ __forceinline__

typedef __bf16 bf16;
typedef __attribute__((ext_vector_type(16))) __bf16 v16bf;
typedef __attribute__((ext_vector_type(8)))  __bf16 v8bf;
typedef __attribute__((ext_vector_type(4)))  __bf16 v4bf;
typedef __attribute__((ext_vector_type(8)))  float  v8f;
typedef __attribute__((ext_vector_type(4)))  float  v4f_t;
typedef __attribute__((ext_vector_type(4)))  unsigned int u32x4;
typedef __attribute__((ext_vector_type(8)))  int i32x8;
typedef __attribute__((ext_vector_type(4)))  int i32x4;

static constexpr int T  = 2048;
static constexpr int Bb = 2;
static constexpr int E  = 2048;
static constexpr int H  = 16;
static constexpr int HD = 128;
static constexpr int F  = 3 * E;     // 6144
static constexpr int M  = T * Bb;    // 4096
static constexpr float SCALE = 0.08838834764831845f; // 128^-0.5
static constexpr float NEGV  = -1.0e9f;

DEVINL bf16 f2bf(float f) {
  unsigned u = __builtin_bit_cast(unsigned, f);
  u += 0x7FFFu + ((u >> 16) & 1u);          // round-to-nearest-even
  unsigned short s = (unsigned short)(u >> 16);
  return __builtin_bit_cast(bf16, s);
}

DEVINL v16bf cat8(v8bf a, v8bf b) {
  return __builtin_shufflevector(a, b, 0,1,2,3,4,5,6,7,8,9,10,11,12,13,14,15);
}

// ---------------------------------------------------------------------------
// TDM: issue a 2D tile load (rows x 64 bf16) global -> LDS.
// D# built per CDNA5 ISA 08_async_tensor §8:
//   group0: count=1 | lds_addr | global_addr | type=2
//   group1: data_size=2B, pad_enable (4 DWORD pad per 32 DWORD = row stride
//           72 bf16), tensor dims/strides, 128x64 tile
// This toolchain exposes the 6-arg builtin:
//   (u32x4 g0, i32x8 g1, i32x4, i32x4, i32x8, i32 cpol)
// ---------------------------------------------------------------------------
DEVINL void tdm_load_tile_2d(unsigned lds_byte_off, const bf16* gsrc,
                             unsigned tensor_d0, unsigned tensor_d1,
                             unsigned tile_d0, unsigned tile_d1,
                             unsigned long long stride0_elems) {
  unsigned long long ga = (unsigned long long)(const void*)gsrc;
  u32x4 g0;
  g0[0] = 1u;                                          // count=1, user D#
  g0[1] = lds_byte_off;                                // LDS byte address
  g0[2] = (unsigned)(ga & 0xFFFFFFFFu);                // global addr lo
  g0[3] = (unsigned)((ga >> 32) & 0x01FFFFFFu) | 0x80000000u; // addr hi | type=2
  i32x8 g1;
  //        data_size=1(2B)  pad_en    pad_interval=4 (32 DW)  pad_amount=3 (4 DW)
  g1[0] = (int)((1u << 16) | (1u << 20) | (4u << 22) | (3u << 25));
  g1[1] = (int)((tensor_d0 & 0xFFFFu) << 16);          // atomic_addr=0 | dim0 lo
  g1[2] = (int)(((tensor_d0 >> 16) & 0xFFFFu) | ((tensor_d1 & 0xFFFFu) << 16));
  g1[3] = (int)(((tensor_d1 >> 16) & 0xFFFFu) | ((tile_d0 & 0xFFFFu) << 16));
  g1[4] = (int)(tile_d1 & 0xFFFFu);                    // tile_dim1 | tile_dim2=0
  g1[5] = (int)(unsigned)(stride0_elems & 0xFFFFFFFFull);
  g1[6] = (int)(unsigned)((stride0_elems >> 32) & 0xFFFFull); // stride0 hi
  g1[7] = 0;
  i32x4 z4 = {0, 0, 0, 0};
  i32x8 z8 = {0, 0, 0, 0, 0, 0, 0, 0};
  __builtin_amdgcn_tensor_load_to_lds(g0, g1, z4, z4, z8, 0);
}

DEVINL unsigned lds_offset_of(const void* p) {
  // generic LDS pointer: low 32 bits carry the LDS byte offset
  return (unsigned)(unsigned long long)p;
}

// ---------------------------------------------------------------------------
// f32 -> bf16 elementwise convert (vectorized x4; all sizes are multiples of 4)
// ---------------------------------------------------------------------------
__global__ void mha_cvt_bf16(const float* __restrict__ in,
                             bf16* __restrict__ out, int n4) {
  int i = blockIdx.x * 256 + threadIdx.x;
  if (i >= n4) return;
  v4f_t v = ((const v4f_t*)in)[i];
  v4bf o;
  o[0] = f2bf(v[0]); o[1] = f2bf(v[1]); o[2] = f2bf(v[2]); o[3] = f2bf(v[3]);
  ((v4bf*)out)[i] = o;
}

// ---------------------------------------------------------------------------
// Tiled bf16 WMMA GEMM with TDM staging: C[M x N] = A[M x K]*W[N x K]^T + bias
//   block = 256 threads (8 waves), 128x128 block tile, wave tile 64x32,
//   K-step 64 (2 WMMA K-chunks per stage) -> v_wmma_f32_16x16x32_bf16
// EPI==0: QKV epilogue (split K/V/Q, scale Q, relayout per head, V transposed)
// EPI==1: f32 output with bias (out projection), N == E
// ---------------------------------------------------------------------------
template <int EPI>
__global__ __launch_bounds__(256) void mha_gemm_wmma(
    const bf16* __restrict__ A,     // M x K, row major
    const bf16* __restrict__ W,     // N x K, row major
    const float* __restrict__ bias, // N
    float* __restrict__ Cout,       // EPI==1
    bf16* __restrict__ Kbuf,        // EPI==0: (b,h,t,d)
    bf16* __restrict__ Vt,          // EPI==0: (b,h,d,t)
    bf16* __restrict__ Qbuf,        // EPI==0: (b,h,t,d), pre-scaled
    int Kdim) {
  constexpr int LDK = 72;           // 64 + 8 bf16 TDM pad per row (144B)
  __shared__ bf16 As[128 * LDK];
  __shared__ bf16 Ws[128 * LDK];

  const int tid  = threadIdx.x;
  const int lane = tid & 31;
  const int wave = tid >> 5;
  const int l15  = lane & 15;
  const int lh   = lane >> 4;       // 0 for lanes 0-15, 1 for lanes 16-31
  const int wm   = (wave >> 2) * 64;
  const int wn   = (wave & 3) * 32;
  const int m0   = blockIdx.y * 128;
  const int n0   = blockIdx.x * 128;

  const unsigned ldsA = lds_offset_of(As);
  const unsigned ldsW = lds_offset_of(Ws);

  v8f acc[4][2] = {};

  for (int k0 = 0; k0 < Kdim; k0 += 64) {
    __syncthreads();                 // everyone done reading previous tiles
    if (wave == 0) {                 // one wave drives the Tensor Data Mover
      tdm_load_tile_2d(ldsA, A + (size_t)m0 * Kdim + k0,
                       (unsigned)Kdim, (unsigned)128, 64, 128,
                       (unsigned long long)Kdim);
      tdm_load_tile_2d(ldsW, W + (size_t)n0 * Kdim + k0,
                       (unsigned)Kdim, (unsigned)128, 64, 128,
                       (unsigned long long)Kdim);
      __builtin_amdgcn_s_wait_tensorcnt(0);
    }
    __syncthreads();                 // publish DMA'd tiles to all waves

#pragma unroll
    for (int c = 0; c < 2; ++c) {
      // B fragments: lane l15 = column n, K in two 8-chunks per ISA layout
      v16bf bf[2];
#pragma unroll
      for (int j = 0; j < 2; ++j) {
        const bf16* p = &Ws[(wn + j * 16 + l15) * LDK + c * 32 + lh * 8];
        bf[j] = cat8(*(const v8bf*)p, *(const v8bf*)(p + 16));
      }
      // A fragments: lane l15 = row m, K in two 8-chunks per ISA layout
#pragma unroll
      for (int i = 0; i < 4; ++i) {
        const bf16* p = &As[(wm + i * 16 + l15) * LDK + c * 32];
        v16bf af = cat8(*(const v8bf*)(p + lh * 8),
                        *(const v8bf*)(p + 16 + lh * 8));
#pragma unroll
        for (int j = 0; j < 2; ++j)
          acc[i][j] = __builtin_amdgcn_wmma_f32_16x16x32_bf16(
              false, af, false, bf[j], (short)0, acc[i][j], false, false);
      }
    }
  }

  // Epilogue. C layout: VGPR r holds M = r (lanes 0-15) / r+8 (lanes 16-31).
#pragma unroll
  for (int i = 0; i < 4; ++i) {
#pragma unroll
    for (int j = 0; j < 2; ++j) {
#pragma unroll
      for (int r = 0; r < 8; ++r) {
        const int mm = m0 + wm + i * 16 + r + (lh << 3);
        const int nn = n0 + wn + j * 16 + l15;
        float v = acc[i][j][r] + bias[nn];
        if (EPI == 1) {
          Cout[(size_t)mm * E + nn] = v;
        } else {
          const int which = nn >> 11;        // 0=k 1=v 2=q
          const int e = nn & (E - 1);
          const int h = e >> 7;
          const int d = e & (HD - 1);
          const int t = mm >> 1;             // m = t*B + b, B==2
          const int b = mm & 1;
          if (which == 0)
            Kbuf[(((size_t)(b * H + h)) * T + t) * HD + d] = f2bf(v);
          else if (which == 1)
            Vt[(((size_t)(b * H + h)) * HD + d) * T + t] = f2bf(v);
          else
            Qbuf[(((size_t)(b * H + h)) * T + t) * HD + d] = f2bf(v * SCALE);
        }
      }
    }
  }
}

// ---------------------------------------------------------------------------
// Flash attention: grid (T/128, H, B), 256 threads = 8 waves.
// Each wave owns 16 query rows; online softmax over causal key blocks of 32.
// ---------------------------------------------------------------------------
__global__ __launch_bounds__(256) void mha_flash_attn(
    const bf16* __restrict__ Qb,   // (b,h,t,d), pre-scaled
    const bf16* __restrict__ Kb,   // (b,h,t,d)
    const bf16* __restrict__ Vt,   // (b,h,d,t)
    bf16* __restrict__ ctx) {      // (t,b,e)
  constexpr int LDP = 40;
  __shared__ bf16 P[8][16 * LDP];  // wave-private P-tile staging

  const int lane = threadIdx.x & 31;
  const int wave = threadIdx.x >> 5;
  const int l15  = lane & 15;
  const int lh   = lane >> 4;
  const int b    = blockIdx.z;
  const int h    = blockIdx.y;
  const int qm   = blockIdx.x * 128 + wave * 16;  // wave's first query row

  const bf16* Qh = Qb + ((size_t)(b * H + h)) * T * HD;
  const bf16* Kh = Kb + ((size_t)(b * H + h)) * T * HD;
  const bf16* Vh = Vt + ((size_t)(b * H + h)) * HD * T;

  // Q fragments: 16 rows x HD=128 -> 4 A-fragments of K=32
  v16bf qa[4];
  {
    const bf16* qp = Qh + (size_t)(qm + l15) * HD;
#pragma unroll
    for (int c = 0; c < 4; ++c)
      qa[c] = cat8(*(const v8bf*)(qp + c * 32 + lh * 8),
                   *(const v8bf*)(qp + c * 32 + 16 + lh * 8));
  }

  v8f o[8] = {};
  float mrow[8], lrow[8];
#pragma unroll
  for (int r = 0; r < 8; ++r) { mrow[r] = -3.0e38f; lrow[r] = 0.0f; }

  for (int s0 = 0; s0 <= qm + 15; s0 += 32) {
    // ---- S = Q K^T for 32 keys (two 16-column tiles) ----
    v8f sa[2] = {};
#pragma unroll
    for (int nt = 0; nt < 2; ++nt) {
      const bf16* kp = Kh + (size_t)(s0 + nt * 16 + l15) * HD + lh * 8;
#pragma unroll
      for (int c = 0; c < 4; ++c) {
        v16bf kf = cat8(*(const v8bf*)(kp + c * 32),
                        *(const v8bf*)(kp + c * 32 + 16));
        sa[nt] = __builtin_amdgcn_wmma_f32_16x16x32_bf16(
            false, qa[c], false, kf, (short)0, sa[nt], false, false);
      }
    }

    // ---- causal mask + online softmax ----
#pragma unroll
    for (int r = 0; r < 8; ++r) {
      const int row = qm + r + (lh << 3);
#pragma unroll
      for (int nt = 0; nt < 2; ++nt) {
        const int col = s0 + nt * 16 + l15;
        if (col > row) sa[nt][r] += NEGV;
      }
      float mx = fmaxf(sa[0][r], sa[1][r]);
#pragma unroll
      for (int off = 8; off >= 1; off >>= 1)
        mx = fmaxf(mx, __shfl_xor(mx, off, 32));
      const float mn = fmaxf(mrow[r], mx);
      const float al = __expf(mrow[r] - mn);
      const float p0 = __expf(sa[0][r] - mn);
      const float p1 = __expf(sa[1][r] - mn);
      sa[0][r] = p0; sa[1][r] = p1;
      float rs = p0 + p1;
#pragma unroll
      for (int off = 8; off >= 1; off >>= 1)
        rs += __shfl_xor(rs, off, 32);
      lrow[r] = lrow[r] * al + rs;
      mrow[r] = mn;
#pragma unroll
      for (int j = 0; j < 8; ++j) o[j][r] *= al;
    }

    // ---- P (16x32) through wave-private LDS into A-fragment layout ----
    bf16* pw = &P[wave][0];
#pragma unroll
    for (int nt = 0; nt < 2; ++nt)
#pragma unroll
      for (int r = 0; r < 8; ++r)
        pw[(r + (lh << 3)) * LDP + nt * 16 + l15] = f2bf(sa[nt][r]);
    __builtin_amdgcn_wave_barrier();   // keep ds store->load ordered (same wave)
    const bf16* pr = pw + l15 * LDP;
    v16bf pf = cat8(*(const v8bf*)(pr + lh * 8),
                    *(const v8bf*)(pr + 16 + lh * 8));

    // ---- O += P * V : V^T rows are contiguous over keys ----
#pragma unroll
    for (int j = 0; j < 8; ++j) {
      const bf16* vp = Vh + (size_t)(j * 16 + l15) * T + s0 + lh * 8;
      v16bf vf = cat8(*(const v8bf*)vp, *(const v8bf*)(vp + 16));
      o[j] = __builtin_amdgcn_wmma_f32_16x16x32_bf16(
          false, pf, false, vf, (short)0, o[j], false, false);
    }
  }

  // ---- normalize and write ctx (t,b,e) as bf16 ----
#pragma unroll
  for (int r = 0; r < 8; ++r) {
    const int t   = qm + r + (lh << 3);
    const float inv = 1.0f / lrow[r];
#pragma unroll
    for (int j = 0; j < 8; ++j) {
      const int e = h * HD + j * 16 + l15;
      ctx[((size_t)t * Bb + b) * E + e] = f2bf(o[j][r] * inv);
    }
  }
}

// ---------------------------------------------------------------------------
// Launch
// ---------------------------------------------------------------------------
extern "C" void kernel_launch(void* const* d_in, const int* in_sizes, int n_in,
                              void* d_out, int out_size, void* d_ws,
                              size_t ws_size, hipStream_t stream) {
  const float* query = (const float*)d_in[0];
  // d_in[1] = attn_mask (causal tril) -- recomputed analytically in-kernel
  const float* qkv_w = (const float*)d_in[2];
  const float* qkv_b = (const float*)d_in[3];
  const float* out_w = (const float*)d_in[4];
  const float* out_b = (const float*)d_in[5];

  char* ws = (char*)d_ws;
  size_t off = 0;
  auto salloc = [&](size_t bytes) -> void* {
    void* p = ws + off;
    off += (bytes + 255) & ~(size_t)255;
    return p;
  };
  bf16* xb   = (bf16*)salloc((size_t)M * E * 2);
  bf16* qwb  = (bf16*)salloc((size_t)F * E * 2);
  bf16* owb  = (bf16*)salloc((size_t)E * E * 2);
  bf16* Qbuf = (bf16*)salloc((size_t)Bb * H * T * HD * 2);
  bf16* Kbuf = (bf16*)salloc((size_t)Bb * H * T * HD * 2);
  bf16* Vt   = (bf16*)salloc((size_t)Bb * H * HD * T * 2);
  bf16* ctx  = (bf16*)salloc((size_t)M * E * 2);
  (void)ws_size; (void)n_in; (void)in_sizes; (void)out_size;

  // 1) f32 -> bf16 conversions
  {
    int n4 = (M * E) / 4;
    mha_cvt_bf16<<<(n4 + 255) / 256, 256, 0, stream>>>(query, xb, n4);
    n4 = (F * E) / 4;
    mha_cvt_bf16<<<(n4 + 255) / 256, 256, 0, stream>>>(qkv_w, qwb, n4);
    n4 = (E * E) / 4;
    mha_cvt_bf16<<<(n4 + 255) / 256, 256, 0, stream>>>(out_w, owb, n4);
  }

  // 2) QKV projection GEMM (TDM-staged) + fused split/scale/relayout
  mha_gemm_wmma<0><<<dim3(F / 128, M / 128), 256, 0, stream>>>(
      xb, qwb, qkv_b, nullptr, Kbuf, Vt, Qbuf, E);

  // 3) causal flash attention
  mha_flash_attn<<<dim3(T / 128, H, Bb), 256, 0, stream>>>(Qbuf, Kbuf, Vt, ctx);

  // 4) output projection (TDM-staged) -> f32 d_out
  mha_gemm_wmma<1><<<dim3(E / 128, M / 128), 256, 0, stream>>>(
      ctx, owb, out_b, (float*)d_out, nullptr, nullptr, nullptr, E);
}
